// ModulationIndex_15367392985652
// MI455X (gfx1250) — matrix operand order repaired and checked by reading, not compiled, not run on hardware
//
#include <hip/hip_runtime.h>
#include <hip/hip_bf16.h>

// ---------------------------------------------------------------------------
// ModulationIndex on gfx1250 (CDNA5, wave32, WMMA)
//
// dist[b,c,p,a,s,n] = sum_t softmax_n(-|circdist(phase[b,c,p,s,t], ctr_n)|/T)
//                           * amp[b,c,a,s,t]
// -> per (b,c,s): GEMM  W(360 x 2048, f16) x A^T(2048 x 30, f16) -> f32
// -> entropy over n, MI, transpose to (B,C,S,P,A)
// ---------------------------------------------------------------------------

typedef __attribute__((ext_vector_type(16))) _Float16 v16h;
typedef __attribute__((ext_vector_type(8)))  _Float16 v8h;
typedef __attribute__((ext_vector_type(8)))  float    v8f;

#define NBINS 18
#define DIM_B 4
#define DIM_C 8
#define DIM_P 20
#define DIM_A 30
#define DIM_S 4
#define DIM_T 2048
#define MROWS (DIM_P * NBINS)      // 360
#define MTILES 23                  // ceil(360/16)
#define WAVES 8
#define KC 128                     // K (time) chunk staged in LDS

__global__ __launch_bounds__(256)
void ModIdx_wmma_kernel(const float* __restrict__ phase,
                        const float* __restrict__ amp,
                        float* __restrict__ dist)
{
    // LDS: amplitude chunk [32][KC] fp16 (rows 30,31 zero-padded),
    //      per-wave weight tile [16][KC] fp16.
    __shared__ __align__(16) _Float16 sAmp[32 * KC];
    __shared__ __align__(16) _Float16 sW[WAVES][16 * KC];

    const int tid    = threadIdx.x;
    const int lane   = tid & 31;
    const int wave   = tid >> 5;
    const int laneHi = (lane >> 4) & 1;
    const int lmod   = lane & 15;

    const int batch = blockIdx.x;                  // 0..127  == ((b*C)+c)*S + s
    const int grp   = blockIdx.y;                  // 0..2
    const int tileIdx = grp * WAVES + wave;
    const bool active = tileIdx < MTILES;
    const int tile  = active ? tileIdx : (MTILES - 1);

    const int b = batch >> 5;
    const int c = (batch >> 2) & 7;
    const int s = batch & 3;

    const float* __restrict__ phBase =
        phase + (size_t)(((b * DIM_C + c) * DIM_P) * DIM_S + s) * DIM_T; // +p*S*T
    const float* __restrict__ ampBase =
        amp   + (size_t)(((b * DIM_C + c) * DIM_A) * DIM_S + s) * DIM_T; // +a*S*T

    const int Mbase = tile * 16;
    const int p0 = Mbase / NBINS;
    int p1 = (Mbase + 15) / NBINS;
    if (p1 > DIM_P - 1) p1 = DIM_P - 1;
    const int np = p1 - p0 + 1;     // 1 or 2 distinct phase indices per tile

    const float PI_F      = 3.14159265358979323846f;
    const float TWO_PI    = 6.28318530717958647692f;
    const float INV_2PI   = 0.15915494309189533577f;
    const float INV_TEMP  = 100.0f;                 // 1/TEMP
    const float BINW      = TWO_PI / (float)NBINS;

    v8f acc0 = {};   // N cols 0..15  (a = 0..15)
    v8f acc1 = {};   // N cols 16..31 (a = 16..29 valid)

    for (int t0 = 0; t0 < DIM_T; t0 += KC) {
        // -------- stage amplitude chunk: fp32 global -> fp16 LDS ----------
        if (t0 + KC < DIM_T)
            __builtin_prefetch(ampBase + t0 + KC, 0, 1);   // global_prefetch_b8
        for (int idx = tid; idx < 32 * KC; idx += 256) {
            int a  = idx >> 7;          // /KC
            int tl = idx & (KC - 1);
            float v = (a < DIM_A) ? ampBase[(size_t)a * (DIM_S * DIM_T) + t0 + tl]
                                  : 0.0f;
            sAmp[a * KC + tl] = (_Float16)v;
        }

        // -------- softmax phase-binning weights for this wave's 16 rows ---
        _Float16* __restrict__ W = sW[wave];
        for (int i = lane; i < np * KC; i += 32) {
            int p  = p0 + i / KC;
            int tl = i & (KC - 1);
            float ph = phBase[(size_t)p * (DIM_S * DIM_T) + t0 + tl];

            float d[NBINS];
            float mn = 1e30f;
            #pragma unroll
            for (int n = 0; n < NBINS; ++n) {
                float ctr = -PI_F + ((float)n + 0.5f) * BINW;
                float x = ph - ctr + PI_F;
                float q = truncf(x * INV_2PI);          // truncated mod (fmod)
                float dd = fabsf(x - q * TWO_PI - PI_F);
                d[n] = dd;
                mn = fminf(mn, dd);
            }
            float sum = 0.0f;
            #pragma unroll
            for (int n = 0; n < NBINS; ++n) {
                float e = __expf((mn - d[n]) * INV_TEMP);
                d[n] = e;
                sum += e;
            }
            float inv = __builtin_amdgcn_rcpf(sum);
            int rbase = p * NBINS - Mbase;
            #pragma unroll
            for (int n = 0; n < NBINS; ++n) {
                int lr = rbase + n;
                if (lr >= 0 && lr < 16)
                    W[lr * KC + tl] = (_Float16)(d[n] * inv);
            }
        }

        __syncthreads();

        // -------- WMMA over this chunk: 4 k-steps of 32 --------------------
        #pragma unroll
        for (int kb = 0; kb < KC; kb += 32) {
            // A fragment (16x32 f16): M = lane%16,
            //   e=0..15 -> K = kb + e + 16*laneHi
            const _Float16* abase = W + lmod * KC + kb + 16 * laneHi;
            v8h alo = *(const v8h*)(abase);
            v8h ahi = *(const v8h*)(abase + 8);
            v16h af = __builtin_shufflevector(alo, ahi,
                0, 1, 2, 3, 4, 5, 6, 7, 8, 9, 10, 11, 12, 13, 14, 15);

            // B fragments (32x16 f16): N = lane%16, e -> K = kb + e + 16*laneHi
            const _Float16* b0 = sAmp + lmod * KC + kb + 16 * laneHi;
            v8h b0lo = *(const v8h*)(b0);
            v8h b0hi = *(const v8h*)(b0 + 8);
            v16h bf0 = __builtin_shufflevector(b0lo, b0hi,
                0, 1, 2, 3, 4, 5, 6, 7, 8, 9, 10, 11, 12, 13, 14, 15);

            const _Float16* b1 = sAmp + (16 + lmod) * KC + kb + 16 * laneHi;
            v8h b1lo = *(const v8h*)(b1);
            v8h b1hi = *(const v8h*)(b1 + 8);
            v16h bf1 = __builtin_shufflevector(b1lo, b1hi,
                0, 1, 2, 3, 4, 5, 6, 7, 8, 9, 10, 11, 12, 13, 14, 15);

            acc0 = __builtin_amdgcn_wmma_f32_16x16x32_f16(
                false, af, false, bf0, (short)0, acc0, false, false);
            acc1 = __builtin_amdgcn_wmma_f32_16x16x32_f16(
                false, af, false, bf1, (short)0, acc1, false, false);
        }

        __syncthreads();   // protect LDS before next chunk overwrites it
    }

    // -------- store dist[batch][M][a] (f32) ------------------------------
    if (active) {
        float* __restrict__ out = dist + (size_t)batch * (MROWS * DIM_A);
        #pragma unroll
        for (int r = 0; r < 8; ++r) {
            int M = Mbase + r + 8 * laneHi;     // C/D layout: VGPR r -> M row
            if (M < MROWS) {
                int col0 = lmod;
                if (col0 < DIM_A) out[M * DIM_A + col0] = acc0[r];
                int col1 = 16 + lmod;
                if (col1 < DIM_A) out[M * DIM_A + col1] = acc1[r];
            }
        }
    }
}

// ---------------------------------------------------------------------------
// Entropy / MI over the 18 bins, output in (B,C,S,P,A) layout.
// ---------------------------------------------------------------------------
__global__ __launch_bounds__(256)
void ModIdx_entropy_kernel(const float* __restrict__ dist,
                           float* __restrict__ out)
{
    int o = blockIdx.x * 256 + threadIdx.x;
    if (o >= DIM_B * DIM_C * DIM_S * DIM_P * DIM_A) return;

    int a = o % DIM_A;
    int p = (o / DIM_A) % DIM_P;
    int s = (o / (DIM_A * DIM_P)) % DIM_S;
    int c = (o / (DIM_A * DIM_P * DIM_S)) % DIM_C;
    int b = o / (DIM_A * DIM_P * DIM_S * DIM_C);
    int batch = (b * DIM_C + c) * DIM_S + s;

    const float* __restrict__ d =
        dist + (size_t)batch * (MROWS * DIM_A) + (size_t)p * NBINS * DIM_A + a;

    float v[NBINS];
    float sum = 0.0f;
    #pragma unroll
    for (int n = 0; n < NBINS; ++n) { v[n] = d[n * DIM_A]; sum += v[n]; }

    float inv = 1.0f / (sum + 1e-10f);
    float ne = 0.0f;
    #pragma unroll
    for (int n = 0; n < NBINS; ++n) {
        float pn = fmaxf(v[n] * inv, 1e-10f);
        ne += pn * __logf(pn);
    }
    out[o] = 1.0f + ne * 0.34594316186372978f;   // 1 / ln(18)
}

extern "C" void kernel_launch(void* const* d_in, const int* in_sizes, int n_in,
                              void* d_out, int out_size, void* d_ws, size_t ws_size,
                              hipStream_t stream) {
    const float* phase = (const float*)d_in[0];   // (4,8,20,4,2048) f32
    const float* amp   = (const float*)d_in[1];   // (4,8,30,4,2048) f32
    float* dist = (float*)d_ws;                   // 128*360*30 f32 = 5.53 MB
    float* out  = (float*)d_out;                  // 76800 f32

    dim3 grid1(DIM_B * DIM_C * DIM_S, 3);         // 128 batches x 3 tile groups
    ModIdx_wmma_kernel<<<grid1, 256, 0, stream>>>(phase, amp, dist);

    int nout = DIM_B * DIM_C * DIM_S * DIM_P * DIM_A;   // 76800
    ModIdx_entropy_kernel<<<(nout + 255) / 256, 256, 0, stream>>>(dist, out);
}
